// _Attention_6760278524112
// MI455X (gfx1250) — matrix operand order repaired and checked by reading, not compiled, not run on hardware
//
#include <hip/hip_runtime.h>

constexpr int B_  = 4;
constexpr int N_  = 4096;
constexpr int IN_DIM = 128;
constexpr int HID = 64;
constexpr int LDS_STRIDE = N_ + 4;   // pad 4 dwords: lane-half rows land on distinct banks

typedef __attribute__((ext_vector_type(16))) __bf16 v16bf;
typedef __attribute__((ext_vector_type(8)))  __bf16 v8bf;
typedef __attribute__((ext_vector_type(8)))  float  v8f;

__device__ __forceinline__ unsigned short f32_to_bf16_rne(float x) {
    union { float f; unsigned u; } c; c.f = x;
    unsigned r = c.u + 0x7FFFu + ((c.u >> 16) & 1u);
    return (unsigned short)(r >> 16);
}

// ---------------------------------------------------------------------------
// Kernel 1: q = s @ Wq, k = s @ Wk in fp32, rounded to bf16.
// One block per (b,n) row: 128 threads, waves 0-1 -> q cols, waves 2-3 -> k.
// W (32 KB each) stays resident in cache; s row staged in LDS.
// ---------------------------------------------------------------------------
__global__ __launch_bounds__(128) void proj_bf16_kernel(
    const float* __restrict__ s,
    const float* __restrict__ qw,
    const float* __restrict__ kw,
    unsigned short* __restrict__ qout,
    unsigned short* __restrict__ kout)
{
    __shared__ float srow[IN_DIM];
    const int row = blockIdx.x;            // 0 .. B*N-1
    const int t   = threadIdx.x;           // 0 .. 127
    srow[t] = s[(size_t)row * IN_DIM + t];
    __syncthreads();

    const bool isQ = (t < HID);            // wave-uniform (waves 0,1 vs 2,3)
    const float* __restrict__ W = isQ ? qw : kw;
    const int m = t & (HID - 1);
    float acc = 0.f;
#pragma unroll 8
    for (int c = 0; c < IN_DIM; ++c)
        acc = fmaf(srow[c], W[c * HID + m], acc);

    unsigned short* dst = isQ ? qout : kout;
    dst[(size_t)row * HID + m] = f32_to_bf16_rne(acc);
}

// ---------------------------------------------------------------------------
// Kernel 2: att = (q k^T)^2 * gmat, row-normalized. One block per 16-row
// strip. 8 waves x 512 columns each; bf16 WMMA (K=64 = 2 x 16x16x32).
// Strip kept in LDS (16 x 4100 f32) -> single global pass over gmat/out.
// ---------------------------------------------------------------------------
__global__ __launch_bounds__(256) void att_norm_kernel(
    const unsigned short* __restrict__ qbf,
    const unsigned short* __restrict__ kbf,
    const float* __restrict__ gmat,
    float* __restrict__ out)
{
    extern __shared__ float lds[];                 // [16*LDS_STRIDE] att strip
    float* rowsum = lds + 16 * LDS_STRIDE;         // [16] row sums

    const int b    = blockIdx.x >> 8;              // / (N/16)
    const int i0   = (blockIdx.x & 255) << 4;      // strip start row
    const int tid  = threadIdx.x;
    const int wave = tid >> 5;
    const int lane = tid & 31;
    const int h    = lane >> 4;                    // lane half (0/1)
    const int c16  = lane & 15;                    // lane within half

    if (tid < 16) rowsum[tid] = 0.f;
    __syncthreads();

    // ---- A operands: q rows i0..i0+15, K=0..63 (resident across the loop) --
    // A 16-bit layout: lane(r,h) holds K kb+16h..kb+16h+15 (matches B)
    const unsigned short* qrow = qbf + (size_t)(b * N_ + i0 + c16) * HID;
    union AOp { v16bf v; v8bf half8[2]; };
    AOp a0, a1;
    a0.half8[0] = *(const v8bf*)(qrow +  0 + h * 16);   // K  0..31 operand
    a0.half8[1] = *(const v8bf*)(qrow +  8 + h * 16);
    a1.half8[0] = *(const v8bf*)(qrow + 32 + h * 16);   // K 32..63 operand
    a1.half8[1] = *(const v8bf*)(qrow + 40 + h * 16);

    v8f rsum = {};
    const size_t kbase = (size_t)(b * N_) * HID;
    const float* gbase = gmat + ((size_t)(b * N_ + i0 + 8 * h)) * N_ + c16;

    const int jstart = wave * (N_ / 8);            // 512 columns per wave
#pragma unroll 1
    for (int jt = 0; jt < (N_ / 8) / 16; ++jt) {
        const int j0 = jstart + jt * 16;

        // B operand: lane half h holds column (j0+c16), K kb+16h..kb+16h+15
        const unsigned short* kp = kbf + kbase + (size_t)(j0 + c16) * HID + h * 16;
        v16bf b0 = *(const v16bf*)(kp);            // K 0..31 slice
        v16bf b1 = *(const v16bf*)(kp + 32);       // K 32..63 slice

        v8f c = {};
        c = __builtin_amdgcn_wmma_f32_16x16x32_bf16(false, a0.v, false, b0,
                                                    (short)0, c, false, false);
        c = __builtin_amdgcn_wmma_f32_16x16x32_bf16(false, a1.v, false, b1,
                                                    (short)0, c, false, false);

        // square, mask, stash in LDS, accumulate row partials
#pragma unroll
        for (int v = 0; v < 8; ++v) {
            float g = gbase[(size_t)v * N_ + j0];
            float a = c[v] * c[v] * g;
            rsum[v] += a;
            lds[(v + 8 * h) * LDS_STRIDE + (j0 + c16)] = a;
        }
    }

    // reduce partials across the 16 lanes of each half, then into LDS rowsum
#pragma unroll
    for (int m = 1; m < 16; m <<= 1) {
#pragma unroll
        for (int v = 0; v < 8; ++v)
            rsum[v] += __shfl_xor(rsum[v], m, 32);
    }
    if (c16 == 0) {
#pragma unroll
        for (int v = 0; v < 8; ++v)
            atomicAdd(&rowsum[v + 8 * h], rsum[v]);
    }
    __syncthreads();

    // normalize strip out of LDS; fully coalesced b32 stores
    float* outb = out + ((size_t)(b * N_ + i0)) * N_;
    for (int idx = tid; idx < 16 * N_; idx += 256) {
        const int r   = idx >> 12;                 // / N_
        const int col = idx & (N_ - 1);
        const float inv = 1.0f / (rowsum[r] + 0.001f);
        outb[(size_t)r * N_ + col] = lds[r * LDS_STRIDE + col] * inv;
    }
}

// ---------------------------------------------------------------------------
extern "C" void kernel_launch(void* const* d_in, const int* in_sizes, int n_in,
                              void* d_out, int out_size, void* d_ws, size_t ws_size,
                              hipStream_t stream)
{
    const float* s    = (const float*)d_in[0];   // [B,N,128]
    const float* gmat = (const float*)d_in[1];   // [B,N,N]
    const float* qw   = (const float*)d_in[2];   // [128,64]
    const float* kw   = (const float*)d_in[3];   // [128,64]
    float* out = (float*)d_out;                  // [B,N,N]

    unsigned short* qbf = (unsigned short*)d_ws;              // 2 MB
    unsigned short* kbf = qbf + (size_t)B_ * N_ * HID;        // 2 MB

    proj_bf16_kernel<<<B_ * N_, 128, 0, stream>>>(s, qw, kw, qbf, kbf);

    const size_t smem = (size_t)(16 * LDS_STRIDE + 16) * sizeof(float); // ~256.3 KB
    att_norm_kernel<<<B_ * (N_ / 16), 256, smem, stream>>>(qbf, kbf, gmat, out);
}
